// SimpleCrossAttention_66992899883331
// MI455X (gfx1250) — matrix operand order repaired and checked by reading, not compiled, not run on hardware
//
#include <hip/hip_runtime.h>
#include <hip/hip_bf16.h>

typedef _Float16 v16h __attribute__((ext_vector_type(16)));
typedef _Float16 v8h  __attribute__((ext_vector_type(8)));
typedef float    v8f  __attribute__((ext_vector_type(8)));
typedef float    v4f  __attribute__((ext_vector_type(4)));

#define D    128
#define BATCH 16
#define SQ   2048
#define SKV  2048
#define WPB  4   // waves per block in attention kernel

static __device__ __forceinline__ v16h cat8(v8h lo, v8h hi) {
  return __builtin_shufflevector(lo, hi, 0,1,2,3,4,5,6,7,8,9,10,11,12,13,14,15);
}

static __device__ __forceinline__ v8h cvt8(const float* p) {
  v4f a = *(const v4f*)p;
  v4f b = *(const v4f*)(p + 4);
  v8f f = __builtin_shufflevector(a, b, 0,1,2,3,4,5,6,7);
  return __builtin_convertvector(f, v8h);
}

static __device__ __forceinline__ v8f wmma_f16(v16h a, v16h b, v8f c) {
  // D = A(16x32 f16) * B(32x16 f16) + C(16x16 f32)
  return __builtin_amdgcn_wmma_f32_16x16x32_f16(false, a, false, b, (short)0, c,
                                                false, false);
}

// ---------------------------------------------------------------------------
// Kernel 1: transpose + convert weights. Wt[w][n][k] = W_w[k][n] in f16.
// ---------------------------------------------------------------------------
__global__ void prep_w(const float* __restrict__ Wq, const float* __restrict__ Wk,
                       const float* __restrict__ Wv, _Float16* __restrict__ Wt) {
  const int w = blockIdx.x;  // 0=Q, 1=K, 2=V
  const float* W = (w == 0) ? Wq : ((w == 1) ? Wk : Wv);
  _Float16* dst = Wt + (size_t)w * D * D;
  for (int idx = threadIdx.x; idx < D * D; idx += blockDim.x) {
    int n = idx / D, k = idx % D;
    dst[idx] = (_Float16)W[k * D + n];
  }
}

// ---------------------------------------------------------------------------
// Kernel 2: projections. One wave computes 16 rows x 128 cols via WMMA.
// z=0: Q = x@Wq+bq -> Qh row-major f16
// z=1: K = y@Wk+bk -> Kh row-major f16
// z=2: V = y@Wv+bv -> VT transposed per batch: VT[b][n][m] f16
// ---------------------------------------------------------------------------
__global__ void __launch_bounds__(128)
proj_kernel(const float* __restrict__ x, const float* __restrict__ y,
            const _Float16* __restrict__ Wt,
            const float* __restrict__ bq, const float* __restrict__ bk,
            const float* __restrict__ bv,
            _Float16* __restrict__ Qh, _Float16* __restrict__ Kh,
            _Float16* __restrict__ VT) {
  const int z = blockIdx.z;
  const float* src  = (z == 0) ? x : y;
  const _Float16* W = Wt + (size_t)z * D * D;
  const float* bias = (z == 0) ? bq : ((z == 1) ? bk : bv);

  const int wave  = threadIdx.x >> 5;
  const int lane  = threadIdx.x & 31;
  const int laneM = lane & 15;
  const int laneHi = lane >> 4;
  const int m0 = (blockIdx.x * (blockDim.x >> 5) + wave) * 16;

  // A fragments: 16x32 f16 tiles of the input rows (fp32 -> f16 in-register).
  // Lane layout (ISA 7.12.2): lanes 0-15: K {0..7,16..23}; lanes 16-31: {8..15,24..31}.
  const float* arow = src + (size_t)(m0 + laneM) * D;
  v16h A[4];
#pragma unroll
  for (int kb = 0; kb < 4; ++kb) {
    const float* p0 = arow + kb * 32 + laneHi * 8;
    const float* p1 = arow + kb * 32 + 16 + laneHi * 8;
    A[kb] = cat8(cvt8(p0), cvt8(p1));
  }

#pragma unroll
  for (int nt = 0; nt < 8; ++nt) {
    v8f acc = {};
    // B fragment from W^T[n][k]: lane holds n = nt*16+laneM, k {0..7,16..23} / {8..15,24..31}.
    const _Float16* wrow = W + (size_t)(nt * 16 + laneM) * D;
#pragma unroll
    for (int kb = 0; kb < 4; ++kb) {
      const v8h* bp = (const v8h*)(wrow + kb * 32 + laneHi * 8);
      v16h B = cat8(bp[0], bp[2]);
      acc = wmma_f16(A[kb], B, acc);
    }
    const float bb = bias[nt * 16 + laneM];  // N depends only on laneM
    if (z < 2) {
      _Float16* dst = (z == 0) ? Qh : Kh;
#pragma unroll
      for (int i = 0; i < 8; ++i) {
        int m = m0 + i + laneHi * 8;  // C layout: VGPR i -> M = i (+8 for hi half)
        dst[(size_t)m * D + nt * 16 + laneM] = (_Float16)(acc[i] + bb);
      }
    } else {
#pragma unroll
      for (int i = 0; i < 8; ++i) {
        int m  = m0 + i + laneHi * 8;
        int b  = m >> 11;          // / SKV
        int ml = m & (SKV - 1);    // % SKV
        VT[(size_t)b * D * SKV + (size_t)(nt * 16 + laneM) * SKV + ml] =
            (_Float16)(acc[i] + bb);
      }
    }
  }
}

// ---------------------------------------------------------------------------
// Kernel 3: flash attention. One wave owns a 16-query tile, iterates 32 keys
// per step: S via WMMA, online softmax, P -> LDS (layout transpose) -> WMMA PV.
// ---------------------------------------------------------------------------
__global__ void __launch_bounds__(WPB * 32)
attn_kernel(const _Float16* __restrict__ Qh, const _Float16* __restrict__ Kh,
            const _Float16* __restrict__ VT, float* __restrict__ out) {
  __shared__ _Float16 lds[WPB][16 * 32];

  const int wave   = threadIdx.x >> 5;
  const int lane   = threadIdx.x & 31;
  const int laneM  = lane & 15;
  const int laneHi = lane >> 4;
  const int b  = blockIdx.y;
  const int qt = blockIdx.x * WPB + wave;
  const size_t qbase = (size_t)b * SQ + (size_t)qt * 16;

  // Q fragments with 1/sqrt(d) folded in.
  const _Float16 scale = (_Float16)0.08838834764831845f;
  v16h Qf[4];
  {
    const _Float16* qrow = Qh + (qbase + laneM) * D;
#pragma unroll
    for (int kb = 0; kb < 4; ++kb) {
      const v8h* p0 = (const v8h*)(qrow + kb * 32 + laneHi * 8);
      const v8h* p1 = (const v8h*)(qrow + kb * 32 + 16 + laneHi * 8);
      v16h q = cat8(*p0, *p1);
      Qf[kb] = q * scale;
    }
  }

  v8f O[8];
#pragma unroll
  for (int i = 0; i < 8; ++i) O[i] = (v8f){};
  v8f mrow, lrow;
#pragma unroll
  for (int i = 0; i < 8; ++i) { mrow[i] = -1e30f; lrow[i] = 0.0f; }

  const _Float16* kbase = Kh + (size_t)b * SKV * D;
  const _Float16* vbase = VT + (size_t)b * D * SKV;
  _Float16* myLds = lds[wave];

  for (int j = 0; j < SKV / 32; ++j) {
    // ---- S = Q K^T (two 16x16 tiles covering 32 keys) ----
    v8f S0 = {}, S1 = {};
    const _Float16* krow0 = kbase + (size_t)(j * 32 + laneM) * D;
    const _Float16* krow1 = krow0 + 16 * D;
#pragma unroll
    for (int kb = 0; kb < 4; ++kb) {
      const v8h* p0 = (const v8h*)(krow0 + kb * 32 + laneHi * 8);
      S0 = wmma_f16(Qf[kb], cat8(p0[0], p0[2]), S0);
      const v8h* p1 = (const v8h*)(krow1 + kb * 32 + laneHi * 8);
      S1 = wmma_f16(Qf[kb], cat8(p1[0], p1[2]), S1);
    }

    // ---- row max over the 32 keys (reduce across 16-lane half) ----
    v8f t;
#pragma unroll
    for (int i = 0; i < 8; ++i) t[i] = fmaxf(S0[i], S1[i]);
#pragma unroll
    for (int off = 1; off < 16; off <<= 1) {
#pragma unroll
      for (int i = 0; i < 8; ++i) t[i] = fmaxf(t[i], __shfl_xor(t[i], off, 32));
    }

    // ---- online softmax update ----
    v8f mnew, alpha;
#pragma unroll
    for (int i = 0; i < 8; ++i) {
      mnew[i]  = fmaxf(mrow[i], t[i]);
      alpha[i] = __expf(mrow[i] - mnew[i]);
      mrow[i]  = mnew[i];
    }
#pragma unroll
    for (int i = 0; i < 8; ++i) {
      S0[i] = __expf(S0[i] - mnew[i]);
      S1[i] = __expf(S1[i] - mnew[i]);
    }
    v8f rs;
#pragma unroll
    for (int i = 0; i < 8; ++i) rs[i] = S0[i] + S1[i];
#pragma unroll
    for (int off = 1; off < 16; off <<= 1) {
#pragma unroll
      for (int i = 0; i < 8; ++i) rs[i] += __shfl_xor(rs[i], off, 32);
    }
#pragma unroll
    for (int i = 0; i < 8; ++i) lrow[i] = lrow[i] * alpha[i] + rs[i];
#pragma unroll
    for (int nt = 0; nt < 8; ++nt) {
#pragma unroll
      for (int i = 0; i < 8; ++i) O[nt][i] *= alpha[i];
    }

    // ---- P (C layout, f32) -> LDS f16 16x32 row-major -> A fragment ----
#pragma unroll
    for (int i = 0; i < 8; ++i) {
      int m = i + laneHi * 8;
      myLds[m * 32 + laneM]      = (_Float16)S0[i];
      myLds[m * 32 + 16 + laneM] = (_Float16)S1[i];
    }
    asm volatile("s_wait_dscnt 0" ::: "memory");  // wave-private LDS transpose
    v16h Pf;
    {
      const v8h* p0 = (const v8h*)(myLds + laneM * 32 + laneHi * 8);
      const v8h* p1 = (const v8h*)(myLds + laneM * 32 + 16 + laneHi * 8);
      Pf = cat8(*p0, *p1);  // DS ops are in-order within a wave: safe vs next stores
    }

    // ---- O += P @ V  (V^T rows give contiguous B fragments) ----
#pragma unroll
    for (int nt = 0; nt < 8; ++nt) {
      const v8h* vp =
          (const v8h*)(vbase + (size_t)(nt * 16 + laneM) * SKV + j * 32 + laneHi * 8);
      O[nt] = wmma_f16(Pf, cat8(vp[0], vp[2]), O[nt]);
    }
  }

  // ---- epilogue: O / l, store fp32 ----
  float* orow = out + qbase * D;
#pragma unroll
  for (int i = 0; i < 8; ++i) {
    float inv = 1.0f / lrow[i];
    int m = i + laneHi * 8;
#pragma unroll
    for (int nt = 0; nt < 8; ++nt)
      orow[(size_t)m * D + nt * 16 + laneM] = O[nt][i] * inv;
  }
}

// ---------------------------------------------------------------------------
extern "C" void kernel_launch(void* const* d_in, const int* in_sizes, int n_in,
                              void* d_out, int out_size, void* d_ws, size_t ws_size,
                              hipStream_t stream) {
  const float* x  = (const float*)d_in[0];
  const float* y  = (const float*)d_in[1];
  const float* Wq = (const float*)d_in[2];
  const float* bq = (const float*)d_in[3];
  const float* Wk = (const float*)d_in[4];
  const float* bk = (const float*)d_in[5];
  const float* Wv = (const float*)d_in[6];
  const float* bv = (const float*)d_in[7];
  float* out = (float*)d_out;

  char* ws = (char*)d_ws;
  _Float16* Wt = (_Float16*)ws;                               // 3*128*128 f16
  _Float16* Qh = (_Float16*)(ws + (size_t)3 * D * D * 2);     // [B*SQ][D] f16
  _Float16* Kh = Qh + (size_t)BATCH * SQ * D;                 // [B*SKV][D] f16
  _Float16* VT = Kh + (size_t)BATCH * SKV * D;                // [B][D][SKV] f16

  prep_w<<<dim3(3), 256, 0, stream>>>(Wq, Wk, Wv, Wt);

  dim3 pgrid(BATCH * SQ / 16 / 4, 1, 3);   // 512 blocks x 4 waves x 3 projections
  proj_kernel<<<pgrid, 128, 0, stream>>>(x, y, Wt, bq, bk, bv, Qh, Kh, VT);

  dim3 agrid(SQ / 16 / WPB, BATCH);        // 32 x 16 blocks, 4 waves each
  attn_kernel<<<agrid, WPB * 32, 0, stream>>>(Qh, Kh, VT, out);
}
